// AdaptiveBlock_13907104105159
// MI455X (gfx1250) — matrix operand order repaired and checked by reading, not compiled, not run on hardware
//
#include <hip/hip_runtime.h>

// ---------------------------------------------------------------------------
// AdaptiveBlock for MI455X (gfx1250).  B=64, T=512, H=1024, K=49, M=B*T=32768.
// All GEMMs run on v_wmma_f32_16x16x32_bf16 (wave32). f32->bf16 conversion is
// fused into LDS staging (v_cvt_pk_bf16_f32); K-loops are double-buffered so
// global_load_b128 for tile k+1 is in flight while WMMAs consume tile k.
// Pipeline:
//   1) gate_gemm : x@Wx^T fused sigmoid*tanh(cells) -> s_t (f32 ws)
//   2) nsmall    : g  = hiddens@Wg^T           (N padded 49->64)
//   3) cv_gemm   : cv = V@Wv^T per batch
//   4) nsmall    : cs = s_t@Ws^T + g
//   5) attn      : z_t, softmax alpha (bf16, zero-padded), beta
//   6) ct_final  : c_t = alpha@V (WMMA) fused beta-blend + residual -> out
// ---------------------------------------------------------------------------

#define DEV __device__ __forceinline__

typedef __bf16 bf16;
typedef bf16  v16bf __attribute__((ext_vector_type(16)));
typedef bf16  v8bf  __attribute__((ext_vector_type(8)));
typedef float v8f   __attribute__((ext_vector_type(8)));

constexpr int BN   = 64;
constexpr int TN   = 512;
constexpr int HN   = 1024;
constexpr int KN   = 49;
constexpr int KPAD = 64;
constexpr int MTOT = BN * TN;      // 32768
constexpr int LDT  = 40;           // padded LDS tile stride (elements)
constexpr int NKT  = HN / 32;      // 32 k-tiles over the H reduction

DEV v8f wmma_bf16(v16bf a, v16bf b, v8f c) {
  return __builtin_amdgcn_wmma_f32_16x16x32_bf16(false, a, false, b,
                                                 (short)0, c, false, false);
}

DEV float fast_rcp(float x)     { return __builtin_amdgcn_rcpf(x); }
DEV float fast_sigmoid(float x) { return fast_rcp(1.0f + __expf(-x)); }
DEV float fast_tanh(float x)    { return 1.0f - 2.0f * fast_rcp(__expf(2.0f * x) + 1.0f); }

// ---- LDS fragment loads (ISA 7.12.2 16-bit operand layouts, wave32) --------
DEV v16bf load_frag_a(const bf16* lds, int mBase, int lane) {
  int row = mBase + (lane & 15);
  int ks  = (lane < 16) ? 0 : 8;
  const bf16* p = lds + row * LDT + ks;
  union { v16bf v; v8bf h[2]; } u;
  u.h[0] = *(const v8bf*)(p);        // K = ks .. ks+7
  u.h[1] = *(const v8bf*)(p + 16);   // K = ks+16 .. ks+23
  return u.v;
}
DEV v16bf load_frag_b(const bf16* lds, int nBase, int lane) {
  int col = nBase + (lane & 15);
  int ks  = (lane < 16) ? 0 : 8;
  const bf16* p = lds + col * LDT + ks;
  union { v16bf v; v8bf h[2]; } u;
  u.h[0] = *(const v8bf*)(p);        // K = ks .. ks+7
  u.h[1] = *(const v8bf*)(p + 16);   // K = ks+16 .. ks+23
  return u.v;
}

// ---- split staging: load phase (regs) / store phase (cvt + LDS) -----------
// OUTER rows x 32 k, float4 chunks; OUTER/32 chunks per thread (256 threads).
template <int OUTER, bool GUARD>
DEV void tile_load(float4 (&r)[OUTER / 32], const float* __restrict__ gbase,
                   long long rowStart, long long gstride, int k0,
                   long long rowLimit, int tid) {
  constexpr int NC = OUTER / 32;
#pragma unroll
  for (int i = 0; i < NC; ++i) {
    int c = tid + i * 256;
    int row = c >> 3, ch = c & 7;
    long long grow = rowStart + row;
    if (!GUARD || grow < rowLimit)
      r[i] = *(const float4*)(gbase + grow * gstride + k0 + ch * 4);
    else
      r[i] = make_float4(0.f, 0.f, 0.f, 0.f);
  }
}
template <int OUTER>
DEV void tile_store(bf16* lds, const float4 (&r)[OUTER / 32], int tid) {
  constexpr int NC = OUTER / 32;
#pragma unroll
  for (int i = 0; i < NC; ++i) {
    int c = tid + i * 256;
    int row = c >> 3, ch = c & 7;
    bf16* d = lds + row * LDT + ch * 4;
    d[0] = (bf16)r[i].x; d[1] = (bf16)r[i].y;
    d[2] = (bf16)r[i].z; d[3] = (bf16)r[i].w;
  }
}

// bf16 global [row][k] -> bf16 LDS row-major (straight 16B copies).
DEV void stage_bf16(bf16* lds, const bf16* gbase, long long rowStart,
                    long long gstride, int k0, int tid) {
#pragma unroll
  for (int i = 0; i < 2; ++i) {       // 128 rows * 4 chunks / 256 threads
    int c = tid + i * 256;
    int row = c >> 2, ch = c & 3;
    const v8bf* s = (const v8bf*)(gbase + (rowStart + row) * gstride + k0 + ch * 8);
    *(v8bf*)(lds + row * LDT + ch * 8) = *s;
  }
}
// f32 global [k][n] (n contiguous) -> bf16 LDS col-major [n][k]; k>=kLimit -> 0.
DEV void stage_f32_T(bf16* lds, const float* gbase, int k0, long long gstride,
                     int n0, int kLimit, int tid) {
#pragma unroll
  for (int i = 0; i < 4; ++i) {       // 32 k-rows x 32 float4 chunks
    int c = tid + i * 256;
    int k = c >> 5, ch = c & 31;
    float4 v = make_float4(0.f, 0.f, 0.f, 0.f);
    if (k0 + k < kLimit)
      v = *(const float4*)(gbase + (long long)(k0 + k) * gstride + n0 + ch * 4);
    lds[(ch * 4 + 0) * LDT + k] = (bf16)v.x;
    lds[(ch * 4 + 1) * LDT + k] = (bf16)v.y;
    lds[(ch * 4 + 2) * LDT + k] = (bf16)v.z;
    lds[(ch * 4 + 3) * LDT + k] = (bf16)v.w;
  }
}

// ---------------------------------------------------------------------------
// K1: gate GEMM [32768x1024]x[1024x1024] + sigmoid*tanh epilogue.
// Block tile 128x128x32, 8 waves (4Mx2N), per-wave 2x4 WMMA tiles.
// Double-buffered LDS; no row guards (grid exactly tiles M and N).
// ---------------------------------------------------------------------------
__global__ __launch_bounds__(256) void gate_gemm_kernel(
    const float* __restrict__ x, const float* __restrict__ Wx,
    const float* __restrict__ cells, float* __restrict__ st) {
  __shared__ bf16 As[2][128 * LDT];
  __shared__ bf16 Bs[2][128 * LDT];
  const int tid = threadIdx.x, lane = tid & 31, wid = tid >> 5;
  const int waveM = wid >> 1, waveN = wid & 1;
  const long long m0 = (long long)blockIdx.x * 128;
  const int       n0 = blockIdx.y * 128;

  v8f acc[2][4];
#pragma unroll
  for (int i = 0; i < 2; ++i)
#pragma unroll
    for (int j = 0; j < 4; ++j) acc[i][j] = (v8f)0.0f;

  float4 ra[4], rb[4];
  tile_load<128, false>(ra, x,  m0, HN, 0, 0, tid);
  tile_load<128, false>(rb, Wx, n0, HN, 0, 0, tid);

  for (int kt = 0; kt < NKT; ++kt) {
    const int cur = kt & 1;
    __syncthreads();
    tile_store<128>(As[cur], ra, tid);
    tile_store<128>(Bs[cur], rb, tid);
    __syncthreads();
    if (kt + 1 < NKT) {               // issue next tile's loads before compute
      tile_load<128, false>(ra, x,  m0, HN, (kt + 1) * 32, 0, tid);
      tile_load<128, false>(rb, Wx, n0, HN, (kt + 1) * 32, 0, tid);
    }
    v16bf af[2], bf_[4];
#pragma unroll
    for (int mi = 0; mi < 2; ++mi)
      af[mi] = load_frag_a(As[cur], waveM * 32 + mi * 16, lane);
#pragma unroll
    for (int ni = 0; ni < 4; ++ni)
      bf_[ni] = load_frag_b(Bs[cur], waveN * 64 + ni * 16, lane);
#pragma unroll
    for (int mi = 0; mi < 2; ++mi)
#pragma unroll
      for (int ni = 0; ni < 4; ++ni)
        acc[mi][ni] = wmma_bf16(af[mi], bf_[ni], acc[mi][ni]);
  }

#pragma unroll
  for (int mi = 0; mi < 2; ++mi)
#pragma unroll
    for (int ni = 0; ni < 4; ++ni)
#pragma unroll
      for (int r = 0; r < 8; ++r) {
        long long m = m0 + waveM * 32 + mi * 16 + r + ((lane < 16) ? 0 : 8);
        int       n = n0 + waveN * 64 + ni * 16 + (lane & 15);
        long long idx = m * HN + n;
        st[idx] = fast_sigmoid(acc[mi][ni][r]) * fast_tanh(cells[idx]);
      }
}

// ---------------------------------------------------------------------------
// K2/K4: narrow GEMM  out[m][0..63] = A[Mx1024] @ W[64x1024]^T (+ addin)
// W rows >= 49 zero-filled. Block tile 128x64x32, 8 waves each 16x64.
// Double-buffered.
// ---------------------------------------------------------------------------
__global__ __launch_bounds__(256) void nsmall_gemm_kernel(
    const float* __restrict__ A, const float* __restrict__ W,
    const float* __restrict__ addin, float* __restrict__ out) {
  __shared__ bf16 As[2][128 * LDT];
  __shared__ bf16 Bs[2][64 * LDT];
  const int tid = threadIdx.x, lane = tid & 31, wid = tid >> 5;
  const long long m0 = (long long)blockIdx.x * 128;

  v8f acc[4];
#pragma unroll
  for (int j = 0; j < 4; ++j) acc[j] = (v8f)0.0f;

  float4 ra[4], rb[2];
  tile_load<128, false>(ra, A, m0, HN, 0, 0, tid);
  tile_load<64, true>(rb, W, 0, HN, 0, KN, tid);

  for (int kt = 0; kt < NKT; ++kt) {
    const int cur = kt & 1;
    __syncthreads();
    tile_store<128>(As[cur], ra, tid);
    tile_store<64>(Bs[cur], rb, tid);
    __syncthreads();
    if (kt + 1 < NKT) {
      tile_load<128, false>(ra, A, m0, HN, (kt + 1) * 32, 0, tid);
      tile_load<64, true>(rb, W, 0, HN, (kt + 1) * 32, KN, tid);
    }
    v16bf af = load_frag_a(As[cur], wid * 16, lane);
#pragma unroll
    for (int ni = 0; ni < 4; ++ni) {
      v16bf bf_ = load_frag_b(Bs[cur], ni * 16, lane);
      acc[ni] = wmma_bf16(af, bf_, acc[ni]);
    }
  }

#pragma unroll
  for (int ni = 0; ni < 4; ++ni)
#pragma unroll
    for (int r = 0; r < 8; ++r) {
      long long m = m0 + wid * 16 + r + ((lane < 16) ? 0 : 8);
      int       n = ni * 16 + (lane & 15);
      long long idx = m * KPAD + n;
      float v = acc[ni][r];
      if (addin) v += addin[idx];
      out[idx] = v;
    }
}

// ---------------------------------------------------------------------------
// K3: cv[b][64][64] = V[b][49x1024] @ Wv[64x1024]^T, one block per batch.
// 8 waves (4Mx2N), per-wave 16x32.  (Tiny: 64 blocks, left single-buffered.)
// ---------------------------------------------------------------------------
__global__ __launch_bounds__(256) void cv_gemm_kernel(
    const float* __restrict__ V, const float* __restrict__ Wv,
    float* __restrict__ cv) {
  __shared__ bf16 As[64 * LDT];
  __shared__ bf16 Bs[64 * LDT];
  const int b = blockIdx.x;
  const int tid = threadIdx.x, lane = tid & 31, wid = tid >> 5;
  const int waveM = wid >> 1, waveN = wid & 1;
  const float* Vb = V + (long long)b * KN * HN;

  v8f acc[2];
  acc[0] = (v8f)0.0f; acc[1] = (v8f)0.0f;

  for (int k0 = 0; k0 < HN; k0 += 32) {
    float4 ra[2], rb[2];
    tile_load<64, true>(ra, Vb, 0, HN, k0, KN, tid);
    tile_load<64, true>(rb, Wv, 0, HN, k0, KN, tid);
    __syncthreads();
    tile_store<64>(As, ra, tid);
    tile_store<64>(Bs, rb, tid);
    __syncthreads();
    v16bf af = load_frag_a(As, waveM * 16, lane);
#pragma unroll
    for (int ni = 0; ni < 2; ++ni) {
      v16bf bf_ = load_frag_b(Bs, waveN * 32 + ni * 16, lane);
      acc[ni] = wmma_bf16(af, bf_, acc[ni]);
    }
  }

#pragma unroll
  for (int ni = 0; ni < 2; ++ni)
#pragma unroll
    for (int r = 0; r < 8; ++r) {
      int m = waveM * 16 + r + ((lane < 16) ? 0 : 8);
      int n = waveN * 32 + ni * 16 + (lane & 15);
      cv[((long long)b * KPAD + m) * KPAD + n] = acc[ni][r];
    }
}

// ---------------------------------------------------------------------------
// K5: attention scores. One 64-thread block per (b,t) row.
//   z_t[k] = sum_j tanh(cv[b][k][j] + g[m][j]) * Wh[j]   (k<49)
//   alpha = softmax(z_t)  (bf16, zero-padded to 64)
//   beta  = extended-softmax tail prob
// ---------------------------------------------------------------------------
__global__ __launch_bounds__(64) void attn_kernel(
    const float* __restrict__ cv, const float* __restrict__ g,
    const float* __restrict__ cs, const float* __restrict__ Wh,
    float* __restrict__ beta, bf16* __restrict__ alphab) {
  const int m = blockIdx.x;        // 0..32767
  const int b = m >> 9;            // T = 512
  const int tid = threadIdx.x;     // 0..63
  __shared__ float grow[KPAD];
  __shared__ float zsh[KPAD];
  __shared__ float red[2];
  __shared__ float zext_sh;

  grow[tid] = g[(long long)m * KPAD + tid];
  __syncthreads();

  float z = -3.0e38f;
  if (tid < KN) {
    const float* cvrow = cv + ((long long)b * KPAD + tid) * KPAD;
    float a = 0.f;
    for (int j = 0; j < KN; ++j) a += fast_tanh(cvrow[j] + grow[j]) * Wh[j];
    z = a;
  }
  zsh[tid] = z;
  if (tid == 63) {                 // spare lane computes z_ext
    const float* csrow = cs + (long long)m * KPAD;
    float a = 0.f;
    for (int j = 0; j < KN; ++j) a += fast_tanh(csrow[j]) * Wh[j];
    zext_sh = a;
  }
  __syncthreads();

  if (tid == 0) {
    float mx = -3.0e38f;
    for (int j = 0; j < KN; ++j) mx = fmaxf(mx, zsh[j]);
    float s = 0.f;
    for (int j = 0; j < KN; ++j) s += __expf(zsh[j] - mx);
    red[0] = mx; red[1] = s;
  }
  __syncthreads();

  float mx = red[0], s = red[1];
  float a = (tid < KN) ? __expf(zsh[tid] - mx) * fast_rcp(s) : 0.f;
  alphab[(long long)m * KPAD + tid] = (bf16)a;

  if (tid == 0) {
    float ze = zext_sh;
    float m2 = fmaxf(mx, ze);
    float et = __expf(ze - m2);
    beta[m] = et * fast_rcp(s * __expf(mx - m2) + et);
  }
}

// ---------------------------------------------------------------------------
// K6: c_t = alpha[b][512x64] @ V[b][64x1024] (K padded, V k>=49 zeroed),
// fused out = beta*s_t + (1-beta)*c_t + hiddens.  Block tile 128x128, K=64.
// ---------------------------------------------------------------------------
__global__ __launch_bounds__(256) void ct_final_kernel(
    const bf16* __restrict__ alphab, const float* __restrict__ V,
    const float* __restrict__ beta, const float* __restrict__ st,
    const float* __restrict__ hiddens, float* __restrict__ out) {
  __shared__ bf16 As[128 * LDT];
  __shared__ bf16 Bs[128 * LDT];
  const int b = blockIdx.x, mt = blockIdx.y, nt = blockIdx.z;
  const int tid = threadIdx.x, lane = tid & 31, wid = tid >> 5;
  const int waveM = wid >> 1, waveN = wid & 1;
  const long long mGlob0 = (long long)b * TN + mt * 128;
  const int n0 = nt * 128;
  const float* Vb = V + (long long)b * KN * HN;

  v8f acc[2][4];
#pragma unroll
  for (int i = 0; i < 2; ++i)
#pragma unroll
    for (int j = 0; j < 4; ++j) acc[i][j] = (v8f)0.0f;

#pragma unroll
  for (int k0 = 0; k0 < KPAD; k0 += 32) {
    __syncthreads();
    stage_bf16(As, alphab, mGlob0, KPAD, k0, tid);
    stage_f32_T(Bs, Vb, k0, HN, n0, KN, tid);   // B[k][n] = V[b][k][n], transposed
    __syncthreads();
    v16bf af[2], bf_[4];
#pragma unroll
    for (int mi = 0; mi < 2; ++mi)
      af[mi] = load_frag_a(As, waveM * 32 + mi * 16, lane);
#pragma unroll
    for (int ni = 0; ni < 4; ++ni)
      bf_[ni] = load_frag_b(Bs, waveN * 64 + ni * 16, lane);
#pragma unroll
    for (int mi = 0; mi < 2; ++mi)
#pragma unroll
      for (int ni = 0; ni < 4; ++ni)
        acc[mi][ni] = wmma_bf16(af[mi], bf_[ni], acc[mi][ni]);
  }

#pragma unroll
  for (int mi = 0; mi < 2; ++mi)
#pragma unroll
    for (int ni = 0; ni < 4; ++ni)
#pragma unroll
      for (int r = 0; r < 8; ++r) {
        long long m = mGlob0 + waveM * 32 + mi * 16 + r + ((lane < 16) ? 0 : 8);
        int       n = n0 + waveN * 64 + ni * 16 + (lane & 15);
        long long idx = m * HN + n;
        float bt = beta[m];
        out[idx] = bt * st[idx] + (1.0f - bt) * acc[mi][ni][r] + hiddens[idx];
      }
}

// ---------------------------------------------------------------------------
extern "C" void kernel_launch(void* const* d_in, const int* in_sizes, int n_in,
                              void* d_out, int out_size, void* d_ws, size_t ws_size,
                              hipStream_t stream) {
  (void)in_sizes; (void)n_in; (void)out_size; (void)ws_size;
  const float* x       = (const float*)d_in[0];
  const float* hiddens = (const float*)d_in[1];
  const float* cells   = (const float*)d_in[2];
  const float* V       = (const float*)d_in[3];
  const float* Wx      = (const float*)d_in[4];
  // d_in[5] = Whh: multiplied by h_prev == 0, unused.
  const float* Wv      = (const float*)d_in[6];
  const float* Wg      = (const float*)d_in[7];
  const float* Ws      = (const float*)d_in[8];
  const float* Wh      = (const float*)d_in[9];
  float* out = (float*)d_out;

  // workspace layout (~156 MB)
  char* w = (char*)d_ws;
  float* st     = (float*)(w);                       // 32768*1024*4 = 134217728
  float* g      = (float*)(w + 134217728);           // 32768*64*4   =   8388608
  float* cs     = (float*)(w + 142606336);           // 32768*64*4   =   8388608
  float* cv     = (float*)(w + 150994944);           // 64*64*64*4   =   1048576
  bf16*  alphab = (bf16*) (w + 152043520);           // 32768*64*2   =   4194304
  float* beta   = (float*)(w + 156237824);           // 32768*4      =    131072

  gate_gemm_kernel<<<dim3(MTOT / 128, HN / 128), 256, 0, stream>>>(x, Wx, cells, st);
  nsmall_gemm_kernel<<<dim3(MTOT / 128), 256, 0, stream>>>(hiddens, Wg, nullptr, g);
  cv_gemm_kernel<<<dim3(BN), 256, 0, stream>>>(V, Wv, cv);
  nsmall_gemm_kernel<<<dim3(MTOT / 128), 256, 0, stream>>>(st, Ws, g, cs);
  attn_kernel<<<dim3(MTOT), 64, 0, stream>>>(cv, g, cs, Wh, beta, alphab);
  ct_final_kernel<<<dim3(BN, TN / 128, HN / 128), 256, 0, stream>>>(
      alphab, V, beta, st, hiddens, out);
}
